// GATv2Net_NodeClassifier_40235253629386
// MI455X (gfx1250) — matrix operand order repaired and checked by reading, not compiled, not run on hardware
//
#include <hip/hip_runtime.h>
#include <hip/hip_bf16.h>

// ---------------------------------------------------------------------------
// GATv2 GNN forward for MI455X (gfx1250, wave32, WMMA).
// Dense GEMMs -> v_wmma_f32_16x16x32_f16 (f32 in/out, f16 compute), K/N
// compile-time so the k-loop fully unrolls; all LDS staging is branch-free
// (clamped addresses + 0/1 masks) so nothing scalarizes.
// Edge softmax/scatter -> f32 global atomics (L2-resident working set).
// ---------------------------------------------------------------------------

#define NN      50000
#define EE      400000
#define IN_F    128
#define HID_F   16
#define HEADS   11
#define HC_F    176          // HID_F * HEADS
#define NGRAPH  64
#define OUT_F   32
#define NC_F    10
#define EPS_GN  1e-5f
#define GN_CHUNK 16

typedef __attribute__((ext_vector_type(16))) _Float16 v16h;
typedef __attribute__((ext_vector_type(8)))  float    v8f;

static __host__ inline int ceil_div(int a, int b) { return (a + b - 1) / b; }

// ---------------------------------------------------------------------------
// GEMM: C[M x NCOL] = act(A[M x K] @ W[K x NCOL] + bias), f16 WMMA compute.
// Block = 256 threads = 8 waves; wave w owns the 16x16 tile at
//   rows [blockIdx.y*16, +16), cols [blockIdx.x*128 + w*16, +16).
// A panel (16x32) and B panel (32x128, stored transposed [col][k], BPAD=40
// so per-lane fragments are 16B-aligned contiguous f16 -> ds_load_b128).
// Staging is branch-free: out-of-range B columns are *clamped* (their
// garbage only reaches never-stored C columns); out-of-range K rows are
// clamped + masked to zero (they would pollute valid accumulators).
// ---------------------------------------------------------------------------
template<int K, int NCOL, bool RELU>
__global__ void gemm_wmma_kernel(const float* __restrict__ A,
                                 const float* __restrict__ W,
                                 const float* __restrict__ bias,
                                 float* __restrict__ C, int M)
{
  constexpr int KSTEPS = (K + 31) / 32;
  constexpr int BPAD   = 40;                    // [col][k] pitch (f16), 16B mult

  __shared__ _Float16 As[16 * 32];              // [row][k]
  __shared__ _Float16 Bs[128 * BPAD];           // [colLocal][k], padded

  const int wave     = threadIdx.x >> 5;
  const int lane     = threadIdx.x & 31;
  const int tm       = blockIdx.y;              // row tile
  const int cb       = blockIdx.x * 128;        // block column base
  const int colLocal = wave * 16 + (lane & 15);
  const int col      = cb + colLocal;

  v8f acc = {};

#pragma unroll
  for (int s = 0; s < KSTEPS; ++s) {
    const int kk = s * 32;

    // ---- stage A panel: one clamped float2 load + mask per thread --------
    {
      int r  = threadIdx.x >> 4;                // 0..15
      int c2 = (threadIdx.x & 15) * 2;          // 0,2,...,30
      int gr = tm * 16 + r; if (gr >= M) gr = M - 1;        // clamp rows
      int gk = kk + c2;
      int gks = gk; if (gks > K - 2) gks = K - 2;           // clamp k (8B align ok)
      float2 v = *(const float2*)&A[(size_t)gr * K + gks];
      float m = (gk + 1 < K) ? 1.0f : 0.0f;                 // K,c2 even -> pair mask
      As[r * 32 + c2]     = (_Float16)(v.x * m);
      As[r * 32 + c2 + 1] = (_Float16)(v.y * m);
    }

    // ---- stage B panel transposed (branch-free) --------------------------
    if constexpr ((NCOL & 3) == 0) {
      // 4x4 micro-transpose: 4 coalesced float4 loads, 4 merged b64 LDS stores
      int c4 = (threadIdx.x & 31) * 4;          // col base 0..124
      int r4 = (threadIdx.x >> 5) * 4;          // k base 0..28
      int gc = cb + c4; if (gc > NCOL - 4) gc = NCOL - 4;   // clamp cols (aligned)
      float4 row[4];
      float  km[4];
#pragma unroll
      for (int j = 0; j < 4; ++j) {
        int gk  = kk + r4 + j;
        int gks = gk; if (gks > K - 1) gks = K - 1;         // clamp k
        row[j]  = *(const float4*)&W[(size_t)gks * NCOL + gc];
        km[j]   = (gk < K) ? 1.0f : 0.0f;
      }
#pragma unroll
      for (int j = 0; j < 4; ++j) {
        _Float16* bp = &Bs[(c4 + j) * BPAD + r4];
        bp[0] = (_Float16)(((const float*)&row[0])[j] * km[0]);
        bp[1] = (_Float16)(((const float*)&row[1])[j] * km[1]);
        bp[2] = (_Float16)(((const float*)&row[2])[j] * km[2]);
        bp[3] = (_Float16)(((const float*)&row[3])[j] * km[3]);
      }
    } else {
      // scalar branch-free fallback (only the tiny 32x10 classifier GEMM)
#pragma unroll
      for (int i = threadIdx.x; i < 4096; i += 256) {
        int r  = i >> 7;                        // k row 0..31
        int c  = i & 127;                       // col local
        int gc = cb + c; if (gc > NCOL - 1) gc = NCOL - 1;
        int gk = kk + r;
        int gks = gk; if (gks > K - 1) gks = K - 1;
        float v = W[(size_t)gks * NCOL + gc] * ((gk < K) ? 1.0f : 0.0f);
        Bs[c * BPAD + r] = (_Float16)v;
      }
    }

    // prefetch next A panel
    if (threadIdx.x < 16 && (kk + 32) < K) {
      int gr = tm * 16 + threadIdx.x; if (gr >= M) gr = M - 1;
      __builtin_prefetch(&A[(size_t)gr * K + kk + 32], 0, 1);
    }
    __syncthreads();

    // ---- A fragment (contiguous k order, same as B) ------------------------------
    // lanes 0-15 : V0..7 -> K 0..15  (row = lane)
    // lanes16-31 : V0..7 -> K 16..31 (row = lane-16)
    v16h a;
    {
      const int row  = lane & 15;
      const int koff = (lane >> 4) * 16;
#pragma unroll
      for (int v = 0; v < 4; ++v) {
        a[2 * v]         = As[row * 32 + koff + 2 * v];
        a[2 * v + 1]     = As[row * 32 + koff + 2 * v + 1];
        a[8 + 2 * v]     = As[row * 32 + 8 + koff + 2 * v];
        a[8 + 2 * v + 1] = As[row * 32 + 8 + koff + 2 * v + 1];
      }
    }

    // ---- B fragment: lane = col; 16 contiguous f16, 16B aligned ----------
    v16h b;
    {
      const int kb = (lane >> 4) * 16;
      const _Float16* bp = &Bs[colLocal * BPAD + kb];
#pragma unroll
      for (int v = 0; v < 16; ++v) b[v] = bp[v];  // -> ds_load_b128 x2
    }

    acc = __builtin_amdgcn_wmma_f32_16x16x32_f16(
        /*neg_a=*/false, a, /*neg_b=*/false, b,
        /*c_mod=*/(short)0, acc, /*reuse_a=*/false, /*reuse_b=*/false);

    __syncthreads();
  }

  // ---- store: lane = col (mod 16); VGPR r -> row r (+8 for hi lanes) -----
  if (col < NCOL) {
    const float bv = bias ? bias[col] : 0.0f;
    const int rbase = tm * 16 + ((lane >> 4) ? 8 : 0);
#pragma unroll
    for (int r = 0; r < 8; ++r) {
      int grow = rbase + r;
      if (grow < M) {
        float v = acc[r] + bv;
        if (RELU) v = fmaxf(v, 0.0f);
        C[(size_t)grow * NCOL + col] = v;
      }
    }
  }
}

template<int K, int NCOL, bool RELU>
static void launch_gemm(const float* A, const float* W, const float* bias,
                        float* C, int M, hipStream_t stream)
{
  dim3 grid(ceil_div(NCOL, 128), ceil_div(M, 16));
  gemm_wmma_kernel<K, NCOL, RELU><<<grid, 256, 0, stream>>>(A, W, bias, C, M);
}

// ---------------------------------------------------------------------------
// Utility fills / counts
// ---------------------------------------------------------------------------
__global__ void fill_kernel(float* __restrict__ p, float v, int n)
{
  int i = blockIdx.x * blockDim.x + threadIdx.x;
  if (i < n) p[i] = v;
}

__global__ void count_nodes_kernel(const int* __restrict__ batch,
                                   float* __restrict__ cnt, int n)
{
  int i = blockIdx.x * blockDim.x + threadIdx.x;
  if (i < n) atomicAdd(&cnt[batch[i]], 1.0f);
}

// ---------------------------------------------------------------------------
// GraphNorm: one-pass sum / sum-of-squares with run-length compressed atomics
// (batch is sorted, so each thread flushes ~1-2 atomics per 16-node chunk).
// ---------------------------------------------------------------------------
__global__ void gn_stats_kernel(const float* __restrict__ x,
                                const int* __restrict__ batch,
                                float* __restrict__ S1, float* __restrict__ S2,
                                int n)
{
  int c = threadIdx.x;
  if (c >= HC_F) return;
  int n0 = blockIdx.x * GN_CHUNK;
  if (n0 >= n) return;
  int nend = n0 + GN_CHUNK; if (nend > n) nend = n;

  int g = batch[n0];
  float s1 = 0.0f, s2 = 0.0f;
  for (int node = n0; node < nend; ++node) {
    int gn = batch[node];
    if (gn != g) {
      atomicAdd(&S1[g * HC_F + c], s1);
      atomicAdd(&S2[g * HC_F + c], s2);
      s1 = 0.0f; s2 = 0.0f; g = gn;
    }
    float v = x[(size_t)node * HC_F + c];
    s1 += v; s2 += v * v;
  }
  atomicAdd(&S1[g * HC_F + c], s1);
  atomicAdd(&S2[g * HC_F + c], s2);
}

// S1 -> mean, S2 -> 1/sqrt(var + eps)  where var = E[(x - ms*mean)^2]
__global__ void gn_finalize_kernel(float* __restrict__ S1, float* __restrict__ S2,
                                   const float* __restrict__ cnt,
                                   const float* __restrict__ ms_, int total)
{
  int i = blockIdx.x * blockDim.x + threadIdx.x;
  if (i >= total) return;
  int g = i / HC_F, c = i % HC_F;
  float cv   = fmaxf(cnt[g], 1.0f);
  float mean = S1[i] / cv;
  float ms   = ms_[c];
  float var  = S2[i] / cv - 2.0f * ms * mean * mean + ms * ms * mean * mean;
  S1[i] = mean;
  S2[i] = rsqrtf(var + EPS_GN);
}

__global__ void gn_apply_kernel(const float* __restrict__ x,
                                const int* __restrict__ batch,
                                const float* __restrict__ mean,
                                const float* __restrict__ inv,
                                const float* __restrict__ w,
                                const float* __restrict__ b,
                                const float* __restrict__ ms,
                                float* __restrict__ out, int total)
{
  int i = blockIdx.x * blockDim.x + threadIdx.x;
  if (i >= total) return;
  int node = i / HC_F, c = i % HC_F;
  int g = batch[node];
  float xc = x[i] - ms[c] * mean[g * HC_F + c];
  float v  = w[c] * xc * inv[g * HC_F + c] + b[c];
  out[i] = fmaxf(v, 0.0f);                    // all GraphNorms feed a ReLU
}

// ---------------------------------------------------------------------------
// GATv2 edge phase. One thread per (edge, head); 16-float gathers as float4.
// ---------------------------------------------------------------------------
__device__ inline void atomic_max_f32(float* addr, float val)
{
  // sign-split trick: nonneg floats order as ints; negatives reversed as uints.
  if (val >= 0.0f) atomicMax((int*)addr, __float_as_int(val));
  else             atomicMin((unsigned int*)addr, __float_as_uint(val));
}

__global__ void edge_alpha_kernel(const float* __restrict__ xl,
                                  const float* __restrict__ xr,
                                  const float* __restrict__ att,
                                  const int* __restrict__ ei,
                                  float* __restrict__ alpha,
                                  float* __restrict__ amax,
                                  int E_, int EP)
{
  int idx = blockIdx.x * blockDim.x + threadIdx.x;
  if (idx >= EP * HEADS) return;
  int e = idx / HEADS, h = idx % HEADS;
  int s = (e < E_) ? ei[e]      : (e - E_);   // self-loops appended
  int d = (e < E_) ? ei[E_ + e] : (e - E_);

  const float4* pl = (const float4*)(xl + (size_t)s * HC_F + h * HID_F);
  const float4* pr = (const float4*)(xr + (size_t)d * HC_F + h * HID_F);
  const float4* pa = (const float4*)(att + h * HID_F);

  float acc = 0.0f;
#pragma unroll
  for (int j = 0; j < HID_F / 4; ++j) {
    float4 l = pl[j], r = pr[j], av = pa[j];
    float v0 = l.x + r.x, v1 = l.y + r.y, v2 = l.z + r.z, v3 = l.w + r.w;
    v0 = (v0 > 0.0f) ? v0 : 0.2f * v0;        // leaky_relu(0.2)
    v1 = (v1 > 0.0f) ? v1 : 0.2f * v1;
    v2 = (v2 > 0.0f) ? v2 : 0.2f * v2;
    v3 = (v3 > 0.0f) ? v3 : 0.2f * v3;
    acc += av.x * v0 + av.y * v1 + av.z * v2 + av.w * v3;
  }
  alpha[idx] = acc;
  atomic_max_f32(&amax[d * HEADS + h], acc);
}

__global__ void edge_exp_kernel(const int* __restrict__ ei,
                                float* __restrict__ alpha,
                                const float* __restrict__ amax,
                                float* __restrict__ den,
                                int E_, int EP)
{
  int idx = blockIdx.x * blockDim.x + threadIdx.x;
  if (idx >= EP * HEADS) return;
  int e = idx / HEADS, h = idx % HEADS;
  int d = (e < E_) ? ei[E_ + e] : (e - E_);
  float am = amax[d * HEADS + h];
  if (am < -1e37f) am = 0.0f;                 // non-finite guard (matches ref)
  float ex = __expf(alpha[idx] - am);
  alpha[idx] = ex;                            // in-place: alpha -> exp
  atomicAdd(&den[d * HEADS + h], ex);
}

__global__ void edge_aggr_kernel(const int* __restrict__ ei,
                                 const float* __restrict__ xl,
                                 const float* __restrict__ alpha,
                                 const float* __restrict__ den,
                                 float* __restrict__ out,
                                 int E_, int EP)
{
  int idx = blockIdx.x * blockDim.x + threadIdx.x;
  if (idx >= EP * HEADS) return;
  int e = idx / HEADS, h = idx % HEADS;
  int s = (e < E_) ? ei[e]      : (e - E_);
  int d = (e < E_) ? ei[E_ + e] : (e - E_);
  float a = alpha[idx] / (den[d * HEADS + h] + 1e-16f);

  const float4* pl = (const float4*)(xl + (size_t)s * HC_F + h * HID_F);
  float*        po = out + (size_t)d * HC_F + h * HID_F;
#pragma unroll
  for (int j = 0; j < HID_F / 4; ++j) {
    float4 l = pl[j];
    atomicAdd(&po[4 * j + 0], l.x * a);
    atomicAdd(&po[4 * j + 1], l.y * a);
    atomicAdd(&po[4 * j + 2], l.z * a);
    atomicAdd(&po[4 * j + 3], l.w * a);
  }
}

// conv5 (concat=False): mean over heads + residual + relu
__global__ void head_mean_kernel(const float* __restrict__ aggr,
                                 const float* __restrict__ res,
                                 float* __restrict__ out, int n)
{
  int i = blockIdx.x * blockDim.x + threadIdx.x;
  if (i >= n * HID_F) return;
  int node = i / HID_F, c = i % HID_F;
  float s = 0.0f;
#pragma unroll
  for (int h = 0; h < HEADS; ++h)
    s += aggr[(size_t)node * HC_F + h * HID_F + c];
  float v = s * (1.0f / HEADS) + res[(size_t)node * HID_F + c];
  out[i] = fmaxf(v, 0.0f);
}

// ---------------------------------------------------------------------------
// Host-side orchestration
// ---------------------------------------------------------------------------
extern "C" void kernel_launch(void* const* d_in, const int* in_sizes, int n_in,
                              void* d_out, int out_size, void* d_ws, size_t ws_size,
                              hipStream_t stream)
{
  (void)n_in; (void)out_size; (void)ws_size;

  const float* x        = (const float*)d_in[0];
  const int*   ei       = (const int*)  d_in[1];
  const int*   batch    = (const int*)  d_in[2];
  const float* W_pre    = (const float*)d_in[3];
  const float* b_pre    = (const float*)d_in[4];
  const float* gn_w     = (const float*)d_in[5];
  const float* gn_b     = (const float*)d_in[6];
  const float* gn_ms    = (const float*)d_in[7];
  const float* conv_Wl  = (const float*)d_in[8];
  const float* conv_Wr  = (const float*)d_in[9];
  const float* conv_att = (const float*)d_in[10];
  const float* conv_b   = (const float*)d_in[11];
  const float* conv_Wres= (const float*)d_in[12];
  const float* c5_Wl    = (const float*)d_in[13];
  const float* c5_Wr    = (const float*)d_in[14];
  const float* c5_att   = (const float*)d_in[15];
  const float* c5_b     = (const float*)d_in[16];
  const float* c5_Wres  = (const float*)d_in[17];
  const float* W_o1     = (const float*)d_in[18];
  const float* b_o1     = (const float*)d_in[19];
  const float* W_o2     = (const float*)d_in[20];
  const float* b_o2     = (const float*)d_in[21];
  const float* W_cls    = (const float*)d_in[22];
  const float* b_cls    = (const float*)d_in[23];

  const int N_ = in_sizes[0] / IN_F;      // 50000
  const int E_ = in_sizes[1] / 2;         // 400000
  const int EP = E_ + N_;                 // edges incl. self-loops

  // workspace layout (floats)
  float* ws   = (float*)d_ws;
  size_t off  = 0;
  float* h    = ws + off; off += (size_t)N_ * HC_F;   // current node features
  float* xl   = ws + off; off += (size_t)N_ * HC_F;
  float* xr   = ws + off; off += (size_t)N_ * HC_F;
  float* agg  = ws + off; off += (size_t)N_ * HC_F;   // residual + aggregation
  float* alph = ws + off; off += (size_t)EP * HEADS;  // alpha, then exp(alpha)
  float* amax = ws + off; off += (size_t)N_ * HEADS;
  float* den  = ws + off; off += (size_t)N_ * HEADS;
  float* S1   = ws + off; off += (size_t)NGRAPH * HC_F;
  float* S2   = ws + off; off += (size_t)NGRAPH * HC_F;
  float* cnt  = ws + off; off += NGRAPH;
  float* mlp  = ws + off; off += (size_t)N_ * OUT_F;  // small MLP scratch

  float* out_cls = (float*)d_out;

  const int T = 256;
  auto fill = [&](float* p, float v, int n) {
    fill_kernel<<<ceil_div(n, T), T, 0, stream>>>(p, v, n);
  };
  auto graphnorm = [&](const float* in, float* out, int layer) {
    fill(S1, 0.0f, NGRAPH * HC_F);
    fill(S2, 0.0f, NGRAPH * HC_F);
    gn_stats_kernel<<<ceil_div(N_, GN_CHUNK), T, 0, stream>>>(in, batch, S1, S2, N_);
    gn_finalize_kernel<<<ceil_div(NGRAPH * HC_F, T), T, 0, stream>>>(
        S1, S2, cnt, gn_ms + layer * HC_F, NGRAPH * HC_F);
    gn_apply_kernel<<<ceil_div(N_ * HC_F, T), T, 0, stream>>>(
        in, batch, S1, S2, gn_w + layer * HC_F, gn_b + layer * HC_F,
        gn_ms + layer * HC_F, out, N_ * HC_F);
  };
  auto edge_phase = [&](const float* att, float* out_aggr) {
    const int tot = EP * HEADS;
    fill(amax, -3.0e38f, N_ * HEADS);
    fill(den, 0.0f, N_ * HEADS);
    edge_alpha_kernel<<<ceil_div(tot, T), T, 0, stream>>>(xl, xr, att, ei, alph, amax, E_, EP);
    edge_exp_kernel  <<<ceil_div(tot, T), T, 0, stream>>>(ei, alph, amax, den, E_, EP);
    edge_aggr_kernel <<<ceil_div(tot, T), T, 0, stream>>>(ei, xl, alph, den, out_aggr, E_, EP);
  };

  // ---- per-graph node counts ----
  fill(cnt, 0.0f, NGRAPH);
  count_nodes_kernel<<<ceil_div(N_, T), T, 0, stream>>>(batch, cnt, N_);

  // ---- pre-layer: h = relu(graphnorm(x @ W_pre + b_pre)) ----
  launch_gemm<IN_F, HC_F, false>(x, W_pre, b_pre, agg, N_, stream);
  graphnorm(agg, h, 0);

  // ---- conv1..conv4 (concat=True) ----
  for (int i = 0; i < 4; ++i) {
    launch_gemm<HC_F, HC_F, false>(h, conv_Wl   + (size_t)i * HC_F * HC_F, nullptr,           xl,  N_, stream);
    launch_gemm<HC_F, HC_F, false>(h, conv_Wr   + (size_t)i * HC_F * HC_F, nullptr,           xr,  N_, stream);
    launch_gemm<HC_F, HC_F, false>(h, conv_Wres + (size_t)i * HC_F * HC_F, conv_b + i * HC_F, agg, N_, stream);
    edge_phase(conv_att + (size_t)i * HEADS * HID_F, agg);  // adds on top of residual
    graphnorm(agg, h, i + 1);
  }

  // ---- conv5 (concat=False -> mean over heads) ----
  launch_gemm<HC_F, HC_F, false>(h, c5_Wl, nullptr, xl, N_, stream);
  launch_gemm<HC_F, HC_F, false>(h, c5_Wr, nullptr, xr, N_, stream);
  fill(agg, 0.0f, N_ * HC_F);
  edge_phase(c5_att, agg);
  launch_gemm<HC_F, HID_F, false>(h, c5_Wres, c5_b, mlp, N_, stream);  // residual (N x 16)
  head_mean_kernel<<<ceil_div(N_ * HID_F, T), T, 0, stream>>>(agg, mlp, h, N_);

  // ---- tail MLP ----
  launch_gemm<HID_F, HID_F, true >(h,   W_o1,  b_o1,  xl,      N_, stream);
  launch_gemm<HID_F, OUT_F, true >(xl,  W_o2,  b_o2,  mlp,     N_, stream);
  launch_gemm<OUT_F, NC_F,  false>(mlp, W_cls, b_cls, out_cls, N_, stream);
}
